// LSTMModel_34737695490711
// MI455X (gfx1250) — compile-verified, matrix operand order checked by reading, not run on hardware
//
#include <hip/hip_runtime.h>

// ---------------------------------------------------------------------------
// Problem constants (from the reference)
// ---------------------------------------------------------------------------
static constexpr int Bb = 64;    // batch
static constexpr int Tt = 512;   // timesteps
static constexpr int Dd = 256;   // input dim
static constexpr int Hh = 1024;  // hidden dim
static constexpr int Cc = 128;   // classes

typedef __attribute__((ext_vector_type(16))) __bf16 v16bf;
typedef __attribute__((ext_vector_type(8)))  float  v8f;

// ---------------------------------------------------------------------------
// Helpers
// ---------------------------------------------------------------------------
__device__ __forceinline__ unsigned short f2bf(float f) {
    union { float f; unsigned u; } v; v.f = f;
    unsigned u = v.u;
    u += 0x7FFFu + ((u >> 16) & 1u);   // round-to-nearest-even
    return (unsigned short)(u >> 16);
}

__device__ __forceinline__ float fast_sigmoid(float x) {
    return 1.0f / (1.0f + __expf(-x));
}
__device__ __forceinline__ float fast_tanh(float x) {
    // 1 - 2/(1+e^{2x}): correct limits at +/-inf, single v_exp
    return 1.0f - 2.0f / (1.0f + __expf(2.0f * x));
}

// Load one 16x32 bf16 fragment half for this lane: two contiguous 16B chunks.
// CDNA5 layout: lanes 0-15 hold K={k0..k0+7, k0+16..k0+23},
//               lanes 16-31 hold K={k0+8..15, k0+24..31}  (khalf = lane>>4).
__device__ __forceinline__ v16bf load_frag(const unsigned short* p0) {
    union { v16bf v; uint4 q[2]; } u;
    u.q[0] = *(const uint4*)(p0);
    u.q[1] = *(const uint4*)(p0 + 16);
    return u.v;
}

// ---------------------------------------------------------------------------
// Small utility kernels (precision conversion / init)
// ---------------------------------------------------------------------------
__global__ void cvt_f32_bf16(const float* __restrict__ in,
                             unsigned short* __restrict__ out, int n) {
    int i = blockIdx.x * blockDim.x + threadIdx.x;
    if (i < n) out[i] = f2bf(in[i]);
}

__global__ void bias_combine(const float* __restrict__ a,
                             const float* __restrict__ b,
                             float* __restrict__ o, int n) {
    int i = blockIdx.x * blockDim.x + threadIdx.x;
    if (i < n) o[i] = a[i] + b[i];
}

__global__ void zero_f32(float* p, int n) {
    int i = blockIdx.x * blockDim.x + threadIdx.x;
    if (i < n) p[i] = 0.0f;
}
__global__ void zero_b16(unsigned short* p, int n) {
    int i = blockIdx.x * blockDim.x + threadIdx.x;
    if (i < n) p[i] = 0;
}

// ---------------------------------------------------------------------------
// GEMM accumulation: acc[m][s] += A_tile(m) * W_tile(sec s), K-loop over K.
//   A: [64, K] bf16, row stride lda           (activations)
//   W: [4H, ldw] bf16, gate row = s*Hh+hc+l15 (weights; W^T columns == W rows,
//      so each B-operand lane reads K-contiguous data)
// Base pointers are hoisted; the K loop is unrolled x2 so alternate k-offsets
// fold into the 24-bit global_load immediate instead of pointer increments.
// ---------------------------------------------------------------------------
__device__ __forceinline__ void gemm_part(v8f (&acc)[4][4],
                                          const unsigned short* __restrict__ A,
                                          int lda, int K,
                                          const unsigned short* __restrict__ W,
                                          int ldw, int hc, int l15, int khalf) {
    const unsigned short* pa[4];
    const unsigned short* pb[4];
#pragma unroll
    for (int m = 0; m < 4; ++m)
        pa[m] = A + (size_t)(m * 16 + l15) * (size_t)lda + khalf * 8;
#pragma unroll
    for (int s = 0; s < 4; ++s)
        pb[s] = W + (size_t)(s * Hh + hc + l15) * (size_t)ldw + khalf * 8;

#pragma unroll 2
    for (int k0 = 0; k0 < K; k0 += 32) {
        v16bf a[4];
#pragma unroll
        for (int m = 0; m < 4; ++m) a[m] = load_frag(pa[m] + k0);
        v16bf b[4];
#pragma unroll
        for (int s = 0; s < 4; ++s) {
            b[s] = load_frag(pb[s] + k0);
            // Unconditional speculative prefetch of the next K block
            // (global_prefetch_b8; harmless one-block over-read at loop end).
            __builtin_prefetch(pb[s] + k0 + 32, 0, 1);
        }
#pragma unroll
        for (int m = 0; m < 4; ++m) {
#pragma unroll
            for (int s = 0; s < 4; ++s) {
                acc[m][s] = __builtin_amdgcn_wmma_f32_16x16x32_bf16(
                    false, a[m], false, b[s], (short)0, acc[m][s], false, false);
            }
        }
    }
}

// ---------------------------------------------------------------------------
// One LSTM layer timestep.
//   grid = 16 WGs, block = 128 (4 waves). Wave w of WG b owns h-columns
//   [b*64 + w*16, +16): it computes gate tiles i,f,g,o (4 sections x 4 M-tiles
//   = 16 WMMA accumulators) for those columns, then applies the cell update
//   entirely in registers (each lane owns its (row, col) elements for all
//   four gates -> no LDS exchange needed).
// ---------------------------------------------------------------------------
__global__ __launch_bounds__(128)
void lstm_step(const unsigned short* __restrict__ inA, int lda, int Kin,
               const unsigned short* __restrict__ Wih,   // [4H, Kin] bf16
               const unsigned short* __restrict__ Whh,   // [4H, H]   bf16
               const unsigned short* __restrict__ hprev, // [B, H]    bf16
               const float*          __restrict__ bias,  // [4H] combined
               float*                __restrict__ cstate,// [B, H] f32 in/out
               unsigned short*       __restrict__ hout)  // [B, H] bf16 out
{
    const int lane  = threadIdx.x & 31;
    const int wave  = threadIdx.x >> 5;
    const int khalf = lane >> 4;
    const int l15   = lane & 15;
    const int hc    = blockIdx.x * 64 + wave * 16;  // h-column base of wave

    v8f acc[4][4];
    const v8f vzero = {0.f, 0.f, 0.f, 0.f, 0.f, 0.f, 0.f, 0.f};
#pragma unroll
    for (int m = 0; m < 4; ++m)
#pragma unroll
        for (int s = 0; s < 4; ++s) acc[m][s] = vzero;

    // input contribution  x_t @ Wih^T   (K = Kin)
    gemm_part(acc, inA, lda, Kin, Wih, Kin, hc, l15, khalf);
    // recurrent contribution  h @ Whh^T (K = H)
    gemm_part(acc, hprev, Hh, Hh, Whh, Hh, hc, l15, khalf);

    // ---- LSTM cell update, fully register-resident per lane ----
    const int   col = hc + l15;
    const float bi = bias[0 * Hh + col];
    const float bf = bias[1 * Hh + col];
    const float bg = bias[2 * Hh + col];
    const float bo = bias[3 * Hh + col];

#pragma unroll
    for (int m = 0; m < 4; ++m) {
#pragma unroll
        for (int r = 0; r < 8; ++r) {
            const int row = m * 16 + r + khalf * 8;   // batch index
            const float iv = fast_sigmoid(acc[m][0][r] + bi);
            const float fv = fast_sigmoid(acc[m][1][r] + bf);
            const float gv = fast_tanh   (acc[m][2][r] + bg);
            const float ov = fast_sigmoid(acc[m][3][r] + bo);
            const size_t idx = (size_t)row * Hh + col;
            const float cn = fv * cstate[idx] + iv * gv;
            cstate[idx] = cn;
            hout[idx]   = f2bf(ov * fast_tanh(cn));
        }
    }
}

// ---------------------------------------------------------------------------
// Final FC: out[64,128] = h1 @ Wfc^T + b_fc.  8 WGs x 1 wave, each wave owns
// one 16-column N-tile and all 4 M-tiles; K = 1024.
// ---------------------------------------------------------------------------
__global__ __launch_bounds__(32)
void fc_kernel(const unsigned short* __restrict__ h1,   // [B, H] bf16
               const unsigned short* __restrict__ Wfc,  // [C, H] bf16
               const float*          __restrict__ bfc,  // [C]
               float*                __restrict__ out)  // [B, C] f32
{
    const int lane  = threadIdx.x & 31;
    const int khalf = lane >> 4;
    const int l15   = lane & 15;
    const int col   = blockIdx.x * 16 + l15;

    v8f acc[4];
    const v8f vzero = {0.f, 0.f, 0.f, 0.f, 0.f, 0.f, 0.f, 0.f};
#pragma unroll
    for (int m = 0; m < 4; ++m) acc[m] = vzero;

    const unsigned short* pb0 = Wfc + (size_t)col * Hh + khalf * 8;
    const unsigned short* pa0[4];
#pragma unroll
    for (int m = 0; m < 4; ++m)
        pa0[m] = h1 + (size_t)(m * 16 + l15) * Hh + khalf * 8;

#pragma unroll 2
    for (int k0 = 0; k0 < Hh; k0 += 32) {
        const v16bf b = load_frag(pb0 + k0);
#pragma unroll
        for (int m = 0; m < 4; ++m) {
            const v16bf a = load_frag(pa0[m] + k0);
            acc[m] = __builtin_amdgcn_wmma_f32_16x16x32_bf16(
                false, a, false, b, (short)0, acc[m], false, false);
        }
    }

    const float bb = bfc[col];
#pragma unroll
    for (int m = 0; m < 4; ++m) {
#pragma unroll
        for (int r = 0; r < 8; ++r) {
            const int row = m * 16 + r + khalf * 8;
            out[(size_t)row * Cc + col] = acc[m][r] + bb;
        }
    }
}

// ---------------------------------------------------------------------------
// Host launch
// ---------------------------------------------------------------------------
extern "C" void kernel_launch(void* const* d_in, const int* in_sizes, int n_in,
                              void* d_out, int out_size, void* d_ws, size_t ws_size,
                              hipStream_t stream) {
    const float* x     = (const float*)d_in[0];
    const float* Wih0f = (const float*)d_in[1];
    const float* Whh0f = (const float*)d_in[2];
    const float* bih0  = (const float*)d_in[3];
    const float* bhh0  = (const float*)d_in[4];
    const float* Wih1f = (const float*)d_in[5];
    const float* Whh1f = (const float*)d_in[6];
    const float* bih1  = (const float*)d_in[7];
    const float* bhh1  = (const float*)d_in[8];
    const float* Wfcf  = (const float*)d_in[9];
    const float* bfc   = (const float*)d_in[10];
    float* out = (float*)d_out;

    // Workspace carve-up (~44 MB total)
    char*  ws  = (char*)d_ws;
    size_t off = 0;
    auto alloc = [&](size_t bytes) -> void* {
        void* p = ws + off;
        off += (bytes + 255) & ~(size_t)255;
        return p;
    };
    unsigned short* xb    = (unsigned short*)alloc((size_t)Bb * Tt * Dd * 2);
    unsigned short* wih0  = (unsigned short*)alloc((size_t)4 * Hh * Dd * 2);
    unsigned short* whh0  = (unsigned short*)alloc((size_t)4 * Hh * Hh * 2);
    unsigned short* wih1  = (unsigned short*)alloc((size_t)4 * Hh * Hh * 2);
    unsigned short* whh1  = (unsigned short*)alloc((size_t)4 * Hh * Hh * 2);
    unsigned short* wfc   = (unsigned short*)alloc((size_t)Cc * Hh * 2);
    float*          bias0 = (float*)alloc((size_t)4 * Hh * 4);
    float*          bias1 = (float*)alloc((size_t)4 * Hh * 4);
    float*          cst   = (float*)alloc((size_t)2 * Bb * Hh * 4);  // c0|c1
    unsigned short* hbuf  = (unsigned short*)alloc((size_t)4 * Bb * Hh * 2); // h0x2|h1x2

    float* c0 = cst;
    float* c1 = cst + (size_t)Bb * Hh;
    unsigned short* h0b[2] = {hbuf, hbuf + (size_t)Bb * Hh};
    unsigned short* h1b[2] = {hbuf + (size_t)2 * Bb * Hh, hbuf + (size_t)3 * Bb * Hh};

    const int TB = 256;
    auto grid = [](int n, int tb) { return (n + tb - 1) / tb; };

    // fp32 -> bf16 staging
    { int n = Bb * Tt * Dd;  cvt_f32_bf16<<<grid(n, TB), TB, 0, stream>>>(x, xb, n); }
    { int n = 4 * Hh * Dd;   cvt_f32_bf16<<<grid(n, TB), TB, 0, stream>>>(Wih0f, wih0, n); }
    { int n = 4 * Hh * Hh;   cvt_f32_bf16<<<grid(n, TB), TB, 0, stream>>>(Whh0f, whh0, n); }
    { int n = 4 * Hh * Hh;   cvt_f32_bf16<<<grid(n, TB), TB, 0, stream>>>(Wih1f, wih1, n); }
    { int n = 4 * Hh * Hh;   cvt_f32_bf16<<<grid(n, TB), TB, 0, stream>>>(Whh1f, whh1, n); }
    { int n = Cc * Hh;       cvt_f32_bf16<<<grid(n, TB), TB, 0, stream>>>(Wfcf, wfc, n); }
    { int n = 4 * Hh;        bias_combine<<<grid(n, TB), TB, 0, stream>>>(bih0, bhh0, bias0, n); }
    { int n = 4 * Hh;        bias_combine<<<grid(n, TB), TB, 0, stream>>>(bih1, bhh1, bias1, n); }
    { int n = 2 * Bb * Hh;   zero_f32<<<grid(n, TB), TB, 0, stream>>>(cst, n); }
    { int n = 4 * Bb * Hh;   zero_b16<<<grid(n, TB), TB, 0, stream>>>(hbuf, n); }

    // Sequential recurrence: 2 layer-step kernels per timestep.
    for (int t = 0; t < Tt; ++t) {
        const int pb = t & 1;
        const int nb = pb ^ 1;
        // Layer 0: input x_t (bf16, row stride T*D), K = D
        lstm_step<<<16, 128, 0, stream>>>(xb + (size_t)t * Dd, Tt * Dd, Dd,
                                          wih0, whh0, h0b[pb], bias0, c0, h0b[nb]);
        // Layer 1: input h0_new, K = H
        lstm_step<<<16, 128, 0, stream>>>(h0b[nb], Hh, Hh,
                                          wih1, whh1, h1b[pb], bias1, c1, h1b[nb]);
    }

    // Final classifier on h1 after last step (written into buffer index 0).
    fc_kernel<<<8, 32, 0, stream>>>(h1b[0], wfc, bfc, out);
}
